// task_pattern_propagation_12249246728409
// MI455X (gfx1250) — compile-verified, matrix operand order checked by reading, not run on hardware
//
#include <hip/hip_runtime.h>
#include <hip/hip_bf16.h>

#define DIMC 512u
#define HEADS 8u
#define HEAD_DIM 64u
#define BATCH 4u
#define SEQ 1024u
#define QK_SCALE 0.125f   // 64^-0.5

typedef __attribute__((ext_vector_type(16))) __bf16 v16bf;
typedef __attribute__((ext_vector_type(8)))  float  v8f;

// K index for element e of a 16x32 bf16 A/B fragment (ISA 7.12.2 layout):
// lanes 0-15 / 16-31 are the two half-wave groups; VGPRs 0..3 hold K 0..15
// (group-split), VGPRs 4..7 hold K 16..31. For e=0..7 and e=8..15 the K
// values are contiguous runs of 8 -> fragment gathers vectorize to 128-bit.
__device__ __forceinline__ unsigned frag_k(unsigned lane, unsigned e) {
  unsigned g = (lane >> 4) & 1u;
  unsigned v = e >> 1;
  unsigned p = e & 1u;
  return (v < 4u) ? (8u * g + 2u * v + p) : (16u + 8u * g + 2u * (v - 4u) + p);
}

__device__ __forceinline__ v8f wmma_bf16(v16bf a, v16bf b, v8f c) {
  return __builtin_amdgcn_wmma_f32_16x16x32_bf16(false, a, false, b, (short)0, c,
                                                 false, false);
}

// Workspace element counts
#define QK_ELEMS ((size_t)2 * 2 * BATCH * HEADS * SEQ * HEAD_DIM)  // q,k per stream
#define VT_ELEMS ((size_t)2 * BATCH * HEADS * HEAD_DIM * SEQ)      // v transposed

// ---------------------------------------------------------------------------
// Kernel 1: QKV projection for both streams (bf16 WMMA GEMM, LDS-staged).
// q,k stored [strm][q/k][b][h][n][d]; v stored TRANSPOSED [strm][b][h][d][n]
// so the PV B-fragments later read contiguous 128-bit runs. QK scale folded
// into q.
// ---------------------------------------------------------------------------
__global__ __launch_bounds__(256)
void qkv_proj_kernel(const float* __restrict__ x1, const float* __restrict__ x2,
                     const float* __restrict__ Wq1, const float* __restrict__ Wq2,
                     unsigned short* __restrict__ ws_raw) {
  __bf16* qk = reinterpret_cast<__bf16*>(ws_raw);
  __bf16* vT = qk + QK_ELEMS;
  const unsigned strm = blockIdx.z;
  const float* x = strm ? x2 : x1;
  const float* W = strm ? Wq2 : Wq1;
  const unsigned col0 = blockIdx.x * 32u;  // of 3C = 1536
  const unsigned row0 = blockIdx.y * 64u;  // of B*N = 4096

  __shared__ __bf16 As[64][32];   // x tile, row-major (row, k)
  __shared__ __bf16 Bs[32][32];   // W tile, row-major (out-col, k)

  const unsigned tid  = threadIdx.x;
  const unsigned lane = tid & 31u;
  const unsigned wave = tid >> 5;
  const unsigned mw = (wave >> 1) * 16u;
  const unsigned nw = (wave & 1u) * 16u;
  const unsigned hi = lane >> 4;
  const unsigned nc = lane & 15u;

  v8f acc = {};
  for (unsigned kk = 0; kk < DIMC; kk += 32u) {
#pragma unroll
    for (unsigned j = 0; j < 8u; ++j) {   // stage A 64x32 f32->bf16 (coalesced)
      unsigned e = tid + j * 256u;
      unsigned r = e >> 5, c = e & 31u;
      As[r][c] = (__bf16)x[(size_t)(row0 + r) * DIMC + kk + c];
    }
#pragma unroll
    for (unsigned j = 0; j < 4u; ++j) {   // stage B row-major: Bs[c][k]
      unsigned e = tid + j * 256u;
      unsigned k = e & 31u, c = e >> 5;
      Bs[c][k] = (__bf16)W[(size_t)(col0 + c) * DIMC + kk + k];
    }
    __syncthreads();
    v16bf af, bf;
#pragma unroll
    for (unsigned e = 0; e < 16u; ++e) {
      unsigned k = frag_k(lane, e);
      af[e] = As[mw + nc][k];      // contiguous in k -> ds_load_b128 x2
      bf[e] = Bs[nw + nc][k];      // contiguous in k -> ds_load_b128 x2
    }
    acc = wmma_bf16(af, bf, acc);
    __syncthreads();
  }

#pragma unroll
  for (unsigned v = 0; v < 8u; ++v) {
    unsigned grow = row0 + mw + v + 8u * hi;
    unsigned gcol = col0 + nw + nc;
    unsigned which = gcol / DIMC;       // power-of-2 -> shift
    unsigned rem = gcol & (DIMC - 1u);
    unsigned h  = rem >> 6;             // / HEAD_DIM
    unsigned dd = rem & (HEAD_DIM - 1u);
    unsigned b = grow >> 10;            // / SEQ
    unsigned n = grow & (SEQ - 1u);
    float val = acc[v];
    if (which == 2u) {
      // v: transposed store [strm][b][h][d][n]
      size_t idx = (((size_t)(strm * BATCH + b) * HEADS + h) * HEAD_DIM + dd)
                   * SEQ + n;
      vT[idx] = (__bf16)val;
    } else {
      if (which == 0u) val *= QK_SCALE;
      size_t idx = ((((size_t)(strm * 2u + which) * BATCH + b) * HEADS + h) * SEQ
                    + n) * HEAD_DIM + dd;
      qk[idx] = (__bf16)val;
    }
  }
}

// ---------------------------------------------------------------------------
// Kernel 2: fused dual-stream attention with head-mixing conv + softmax.
// One block per (b, 16-query tile); wave w computes logits for input heads
// (stream1,w) and (stream2,w), shares them through LDS, then each wave owns
// output head h=w: mix -> online softmax -> P@V for both streams.
// ---------------------------------------------------------------------------
__global__ __launch_bounds__(256)
void attn_fused_kernel(const unsigned short* __restrict__ ws_raw,
                       const float* __restrict__ Wconv,
                       const float* __restrict__ bconv,
                       float* __restrict__ o_ws) {
  const __bf16* qk = reinterpret_cast<const __bf16*>(ws_raw);
  const __bf16* vT = qk + QK_ELEMS;
  const unsigned n0 = blockIdx.x * 16u;
  const unsigned b  = blockIdx.y;
  const unsigned wave = threadIdx.x >> 5;
  const unsigned lane = threadIdx.x & 31u;
  const unsigned h  = wave;
  const unsigned hi = lane >> 4;
  const unsigned nc = lane & 15u;

  // S_lds[row][col][head]: the 16-head mix reduction reads 64 contiguous bytes
  __shared__ float  S_lds[16][32][16];
  __shared__ __bf16 P_lds[8][16][32];    // per-wave P relayout scratch

  float wc[16];
#pragma unroll
  for (unsigned c = 0; c < 16u; ++c) wc[c] = Wconv[h * 16u + c];
  const float bias = bconv[h];

  // Q fragments: [stream][K-half of d=64]
  v16bf qf[2][2];
#pragma unroll
  for (unsigned s = 0; s < 2u; ++s) {
    const __bf16* qp = qk +
        ((((size_t)(s * 2u) * BATCH + b) * HEADS + h) * SEQ + n0) * HEAD_DIM;
#pragma unroll
    for (unsigned f = 0; f < 2u; ++f)
#pragma unroll
      for (unsigned e = 0; e < 16u; ++e)
        qf[s][f][e] = qp[(size_t)nc * HEAD_DIM + f * 32u + frag_k(lane, e)];
  }

  const __bf16* kb[2];
  const __bf16* vbT[2];
#pragma unroll
  for (unsigned s = 0; s < 2u; ++s) {
    kb[s]  = qk + (((size_t)(s * 2u + 1u) * BATCH + b) * HEADS + h) * SEQ * HEAD_DIM;
    vbT[s] = vT + (((size_t)s * BATCH + b) * HEADS + h) * HEAD_DIM * SEQ;
  }

  v8f o1[4] = {{0}}, o2[4] = {{0}};
  float row_max[8], row_sum[8];
#pragma unroll
  for (unsigned i = 0; i < 8u; ++i) { row_max[i] = -1e30f; row_sum[i] = 0.0f; }

  for (unsigned m0 = 0; m0 < SEQ; m0 += 32u) {
    if (m0 + 32u < SEQ) {   // hint next K/V tiles toward the caches
      __builtin_prefetch(kb[0] + (size_t)(m0 + 32u) * HEAD_DIM, 0, 1);
      __builtin_prefetch(vbT[0] + (size_t)(m0 + 32u), 0, 1);
    }
    // ---- logits S = q . k^T for this wave's head pair -> LDS ----
#pragma unroll
    for (unsigned s = 0; s < 2u; ++s) {
#pragma unroll
      for (unsigned kt = 0; kt < 2u; ++kt) {
        v8f sacc = {};
#pragma unroll
        for (unsigned ks = 0; ks < 2u; ++ks) {
          v16bf kf;
#pragma unroll
          for (unsigned e = 0; e < 16u; ++e)  // contiguous -> global b128 x2
            kf[e] = kb[s][(size_t)(m0 + kt * 16u + nc) * HEAD_DIM + ks * 32u +
                          frag_k(lane, e)];
          sacc = wmma_bf16(qf[s][ks], kf, sacc);
        }
#pragma unroll
        for (unsigned v = 0; v < 8u; ++v)
          S_lds[v + 8u * hi][kt * 16u + nc][s * 8u + h] = sacc[v];
      }
    }
    __syncthreads();

    // ---- mix 16 -> 1 head with Wconv + bias (contiguous 16-float reads) ----
    v8f mt[2];
#pragma unroll
    for (unsigned t = 0; t < 2u; ++t)
#pragma unroll
      for (unsigned v = 0; v < 8u; ++v) {
        float a = bias;
#pragma unroll
        for (unsigned c = 0; c < 16u; ++c)
          a += wc[c] * S_lds[v + 8u * hi][t * 16u + nc][c];
        mt[t][v] = a;
      }

    // ---- online softmax over this 32-key chunk ----
#pragma unroll
    for (unsigned v = 0; v < 8u; ++v) {
      float rmax = fmaxf(mt[0][v], mt[1][v]);
#pragma unroll
      for (int m = 8; m >= 1; m >>= 1)
        rmax = fmaxf(rmax, __shfl_xor(rmax, m, 32));
      float nmax  = fmaxf(row_max[v], rmax);
      float scale = __expf(row_max[v] - nmax);
      float p0 = __expf(mt[0][v] - nmax);
      float p1 = __expf(mt[1][v] - nmax);
      float psum = p0 + p1;
#pragma unroll
      for (int m = 8; m >= 1; m >>= 1) psum += __shfl_xor(psum, m, 32);
      row_sum[v] = row_sum[v] * scale + psum;
      row_max[v] = nmax;
      mt[0][v] = p0;
      mt[1][v] = p1;
#pragma unroll
      for (unsigned dt = 0; dt < 4u; ++dt) { o1[dt][v] *= scale; o2[dt][v] *= scale; }
    }

    // ---- relayout P (C-layout) -> A-fragment layout via wave-private LDS ----
#pragma unroll
    for (unsigned t = 0; t < 2u; ++t)
#pragma unroll
      for (unsigned v = 0; v < 8u; ++v)
        P_lds[h][v + 8u * hi][t * 16u + nc] = (__bf16)mt[t][v];

    v16bf pf;
#pragma unroll
    for (unsigned e = 0; e < 16u; ++e)    // contiguous -> ds_load_b128 x2
      pf[e] = P_lds[h][nc][frag_k(lane, e)];

    // ---- o += P @ V for both streams (V transposed: contiguous b128) ----
#pragma unroll
    for (unsigned s = 0; s < 2u; ++s)
#pragma unroll
      for (unsigned dt = 0; dt < 4u; ++dt) {
        v16bf vf;
#pragma unroll
        for (unsigned e = 0; e < 16u; ++e)
          vf[e] = vbT[s][(size_t)(dt * 16u + nc) * SEQ + m0 + frag_k(lane, e)];
        if (s == 0u) o1[dt] = wmma_bf16(pf, vf, o1[dt]);
        else         o2[dt] = wmma_bf16(pf, vf, o2[dt]);
      }
    __syncthreads();
  }

  // ---- normalize and store o[b,n, h*64+d] (f32) for both streams ----
#pragma unroll
  for (unsigned v = 0; v < 8u; ++v) {
    float inv = 1.0f / row_sum[v];
    unsigned n = n0 + v + 8u * hi;
#pragma unroll
    for (unsigned dt = 0; dt < 4u; ++dt) {
      size_t idx = ((size_t)b * SEQ + n) * DIMC + h * HEAD_DIM + dt * 16u + nc;
      o_ws[idx] = o1[dt][v] * inv;
      o_ws[(size_t)BATCH * SEQ * DIMC + idx] = o2[dt][v] * inv;
    }
  }
}

// ---------------------------------------------------------------------------
// Kernel 3: output projection y = o @ W^T + b (per stream) -> d_out (f32).
// ---------------------------------------------------------------------------
__global__ __launch_bounds__(256)
void out_proj_kernel(const float* __restrict__ o_ws,
                     const float* __restrict__ W1, const float* __restrict__ b1,
                     const float* __restrict__ W2, const float* __restrict__ b2,
                     float* __restrict__ out) {
  const unsigned strm = blockIdx.z;
  const float* o    = o_ws + (size_t)strm * BATCH * SEQ * DIMC;
  const float* W    = strm ? W2 : W1;
  const float* bias = strm ? b2 : b1;
  float* y = out + (size_t)strm * BATCH * SEQ * DIMC;
  const unsigned col0 = blockIdx.x * 32u;
  const unsigned row0 = blockIdx.y * 64u;

  __shared__ __bf16 As[64][32];
  __shared__ __bf16 Bs[32][32];

  const unsigned tid  = threadIdx.x;
  const unsigned lane = tid & 31u;
  const unsigned wave = tid >> 5;
  const unsigned mw = (wave >> 1) * 16u;
  const unsigned nw = (wave & 1u) * 16u;
  const unsigned hi = lane >> 4;
  const unsigned nc = lane & 15u;

  v8f acc = {};
  for (unsigned kk = 0; kk < DIMC; kk += 32u) {
#pragma unroll
    for (unsigned j = 0; j < 8u; ++j) {
      unsigned e = tid + j * 256u;
      unsigned r = e >> 5, c = e & 31u;
      As[r][c] = (__bf16)o[(size_t)(row0 + r) * DIMC + kk + c];
    }
#pragma unroll
    for (unsigned j = 0; j < 4u; ++j) {
      unsigned e = tid + j * 256u;
      unsigned k = e & 31u, c = e >> 5;
      Bs[c][k] = (__bf16)W[(size_t)(col0 + c) * DIMC + kk + k];
    }
    __syncthreads();
    v16bf af, bf;
#pragma unroll
    for (unsigned e = 0; e < 16u; ++e) {
      unsigned k = frag_k(lane, e);
      af[e] = As[mw + nc][k];
      bf[e] = Bs[nw + nc][k];
    }
    acc = wmma_bf16(af, bf, acc);
    __syncthreads();
  }

  const unsigned gcol = col0 + nw + nc;
  const float bv = bias[gcol];
#pragma unroll
  for (unsigned v = 0; v < 8u; ++v) {
    unsigned grow = row0 + mw + v + 8u * hi;
    y[(size_t)grow * DIMC + gcol] = acc[v] + bv;
  }
}

// ---------------------------------------------------------------------------
extern "C" void kernel_launch(void* const* d_in, const int* in_sizes, int n_in,
                              void* d_out, int out_size, void* d_ws, size_t ws_size,
                              hipStream_t stream) {
  (void)in_sizes; (void)n_in; (void)out_size; (void)ws_size;
  const float* x1    = (const float*)d_in[0];
  const float* x2    = (const float*)d_in[1];
  const float* Wqkv1 = (const float*)d_in[2];
  const float* Wqkv2 = (const float*)d_in[3];
  const float* Wconv = (const float*)d_in[4];
  const float* bconv = (const float*)d_in[5];
  const float* W1    = (const float*)d_in[6];
  const float* b1    = (const float*)d_in[7];
  const float* W2    = (const float*)d_in[8];
  const float* b2    = (const float*)d_in[9];
  float* out = (float*)d_out;

  // Workspace: [qk bf16 | vT bf16 | o f32]
  unsigned short* qkv = (unsigned short*)d_ws;
  const size_t qkv_bytes = (QK_ELEMS + VT_ELEMS) * sizeof(unsigned short);
  float* o_ws = (float*)((char*)d_ws + qkv_bytes);

  qkv_proj_kernel<<<dim3(48, 64, 2), 256, 0, stream>>>(x1, x2, Wqkv1, Wqkv2, qkv);
  attn_fused_kernel<<<dim3(SEQ / 16, BATCH, 1), 256, 0, stream>>>(qkv, Wconv,
                                                                  bconv, o_ws);
  out_proj_kernel<<<dim3(16, 64, 2), 256, 0, stream>>>(o_ws, W1, b1, W2, b2, out);
}
